// LongTermAttention_27169963114768
// MI455X (gfx1250) — matrix operand order repaired and checked by reading, not compiled, not run on hardware
//
#include <hip/hip_runtime.h>

// ---------------------------------------------------------------------------
// LongTermAttention (∞-former continuous attention), analytically simplified:
//   G[l,n] = (2/9) * [l>>2 == n]          (F F^T = 4I exactly for this config)
//   grid softmax == weighted 512-bin softmax with trapezoid bin weights W_n
//                   plus sink term (dt/2)*exp(-m) from the zero-score point t=1
// GEMMs in f16 WMMA (v_wmma_f32_16x16x32_f16), f32 accumulate, f32 output.
// ---------------------------------------------------------------------------

typedef __attribute__((ext_vector_type(16))) _Float16 v16h;
typedef __attribute__((ext_vector_type(8)))  _Float16 h8;
typedef __attribute__((ext_vector_type(8)))  float    v8f;

#define NBATCH 16
#define EDIM   512
#define LDIM   2048
#define NBAS   512
#define TQ     256
#define NHEAD  8
#define DHEAD  64
#define GSCALE (2.0f / 9.0f)
#define DTQ    (1.0f / 999.0f)
#define QSCALE 0.125f

// ---- WMMA helpers (layouts per CDNA5 ISA 7.12.2, wave32) -------------------

__device__ __forceinline__ v16h frag_a(const _Float16* base, int rs, int lane) {
  // A 16x32 f16: lane holds row M=lane&15; halfs 0..7 = K k+0..7 (lane<16) or
  // k+8..15 (lane>=16); halfs 8..15 = +16 of that.
  const _Float16* p = base + (size_t)(lane & 15) * rs + ((lane >> 4) << 3);
  h8 lo = *(const h8*)p;
  h8 hi = *(const h8*)(p + 16);
  return __builtin_shufflevector(lo, hi, 0, 1, 2, 3, 4, 5, 6, 7,
                                         8, 9, 10, 11, 12, 13, 14, 15);
}

__device__ __forceinline__ v16h frag_b(const _Float16* bt, int rs, int lane) {
  // B 32x16 f16 loaded from B^T-row-major ([Ncol][K] contiguous) source:
  // lane holds col N=lane&15; halfs 0..7 = K k+0..7 (lane<16) or k+8..15
  // (lane>=16); halfs 8..15 = +16 of that.
  const _Float16* p = bt + (size_t)(lane & 15) * rs + ((lane >> 4) << 3);
  h8 lo = *(const h8*)p;
  h8 hi = *(const h8*)(p + 16);
  return __builtin_shufflevector(lo, hi, 0, 1, 2, 3, 4, 5, 6, 7,
                                         8, 9, 10, 11, 12, 13, 14, 15);
}

__device__ __forceinline__ v8f wmma_f16(v16h a, v16h b, v8f c) {
  return __builtin_amdgcn_wmma_f32_16x16x32_f16(false, a, false, b,
                                                (short)0, c, false, false);
}

// ---- Prep 1: Bc[b][n][e] = (2/9) * sum_{j<4} k[b][e][4n+j]  (f16) ----------
// LDS-transposed so k reads are 512B-contiguous and Bc writes are coalesced.

__global__ void prep_bc(const float* __restrict__ k, _Float16* __restrict__ bc) {
  __shared__ float tile[32][65];
  int blk = blockIdx.x;                 // 16 batches * 8 nblk * 16 eblk = 2048
  int b  = blk >> 7;
  int t  = blk & 127;
  int n0 = (t >> 4) << 6;               // 64 basis per tile
  int e0 = (t & 15) << 5;               // 32 channels per tile
  int tid = threadIdx.x;
  int nl = tid & 63, ep = tid >> 6;
#pragma unroll
  for (int pass = 0; pass < 8; ++pass) {
    int el = ep + pass * 4;
    const float4 v = *(const float4*)(k + ((size_t)b * EDIM + (e0 + el)) * LDIM
                                        + 4 * (n0 + nl));
    tile[el][nl] = GSCALE * (v.x + v.y + v.z + v.w);
  }
  __syncthreads();
  int el2 = tid & 31, np = tid >> 5;
#pragma unroll
  for (int pass = 0; pass < 8; ++pass) {
    int nl2 = np + pass * 8;
    bc[((size_t)b * NBAS + (n0 + nl2)) * EDIM + e0 + el2] = (_Float16)tile[el2][nl2];
  }
}

// ---- Prep 2: qh[b][h][t][d] = q[b][t][h*64+d] / sqrt(64)  (f16) ------------

__global__ void prep_qh(const float* __restrict__ q, _Float16* __restrict__ qh) {
  int idx = blockIdx.x * 256 + threadIdx.x;   // 16*8*256*64 = 2097152
  int d = idx & 63;
  int t = (idx >> 6) & 255;
  int h = (idx >> 14) & 7;
  int b = idx >> 17;
  qh[idx] = (_Float16)(q[((size_t)b * TQ + t) * EDIM + h * DHEAD + d] * QSCALE);
}

// ---- Prep 3: WkvT[f][e] = (f<512 ? Wk[e][f] : Wv[e][f-512])  (f16) ---------

__global__ void prep_w(const float* __restrict__ Wk, const float* __restrict__ Wv,
                       _Float16* __restrict__ wkvt) {
  int idx = blockIdx.x * 256 + threadIdx.x;   // 1024*512 = 524288
  int e = idx & 511;
  int f = idx >> 9;
  float v = (f < 512) ? Wk[(size_t)e * 512 + f] : Wv[(size_t)e * 512 + (f - 512)];
  wkvt[idx] = (_Float16)v;
}

// ---- GEMM 1: per batch  C[n][f] = Bc[n][:] @ Wkv  (512x512x1024, f32 acc) --
// f<512  -> keys[b][h][n][d]   (natural, strided b16 stores)
// f>=512 -> valsT[b][h][d][n]  (transposed = contiguous b128 stores)

__global__ void gemm_kv(const _Float16* __restrict__ bc,
                        const _Float16* __restrict__ wkvt,
                        _Float16* __restrict__ keys,
                        _Float16* __restrict__ valsT) {
  int lane = threadIdx.x & 31;
  int wid  = threadIdx.x >> 5;
  int w  = blockIdx.x * 8 + wid;        // 4096 waves, 32x64 tile each
  int b  = w >> 8;
  int rm = w & 255;
  int m0 = (rm >> 4) << 5;              // basis-row tile (M)
  int n0 = (rm & 15) << 6;              // feature-col tile (N)
  const _Float16* A = bc + (size_t)b * NBAS * EDIM;
  v8f acc[2][4] = {};
  for (int kk = 0; kk < EDIM; kk += 32) {
    v16h a0 = frag_a(A + (size_t)m0 * EDIM + kk, EDIM, lane);
    v16h a1 = frag_a(A + (size_t)(m0 + 16) * EDIM + kk, EDIM, lane);
#pragma unroll
    for (int j = 0; j < 4; ++j) {
      v16h bf = frag_b(wkvt + (size_t)(n0 + 16 * j) * EDIM + kk, EDIM, lane);
      acc[0][j] = wmma_f16(a0, bf, acc[0][j]);
      acc[1][j] = wmma_f16(a1, bf, acc[1][j]);
    }
  }
#pragma unroll
  for (int i = 0; i < 2; ++i)
#pragma unroll
    for (int j = 0; j < 4; ++j) {
      int f  = n0 + 16 * j + (lane & 15);
      int nb = m0 + 16 * i + ((lane >> 4) << 3);
      if (f < 512) {                              // uniform per (wave,j)
        int h = f >> 6, d = f & 63;
        _Float16* dst = keys + (((size_t)b * NHEAD + h) * NBAS + nb) * DHEAD + d;
#pragma unroll
        for (int rr = 0; rr < 8; ++rr)
          dst[(size_t)rr * DHEAD] = (_Float16)acc[i][j][rr];
      } else {
        int fv = f - 512, h = fv >> 6, d = fv & 63;
        h8 pk;
#pragma unroll
        for (int rr = 0; rr < 8; ++rr) pk[rr] = (_Float16)acc[i][j][rr];
        *(h8*)(valsT + (((size_t)b * NHEAD + h) * DHEAD + d) * NBAS + nb) = pk;
      }
    }
}

// ---- GEMM 2 + weighted softmax:  S = qh @ keys^T  (16 rows x 512 per WG) ---
// m = max(max_n S, 0);  e_n = exp(S-m)*W_n;  Z = sum e_n + (dt/2)exp(-m)
// W_n = dt*count_n - (n==0)*dt/2 ; count_n = ceil(999(n+1)/512)-ceil(999n/512)

__global__ void attn_scores(const _Float16* __restrict__ qh,
                            const _Float16* __restrict__ keys,
                            _Float16* __restrict__ rmat) {
  __shared__ float Sl[16][512];
  __shared__ float red[16][17];
  __shared__ float rowM[16];
  __shared__ float rowZ[16];
  int lane = threadIdx.x & 31;
  int wid  = threadIdx.x >> 5;
  int blk  = blockIdx.x;                // 16*8*16 = 2048
  int t0   = (blk & 15) << 4;
  int bh   = blk >> 4;
  const _Float16* Aq = qh + ((size_t)bh * TQ + t0) * DHEAD;
  const _Float16* Bk = keys + (size_t)bh * NBAS * DHEAD;
  int n0 = wid << 6;                    // this wave: 16 rows x 64 cols
  v8f acc[4] = {};
#pragma unroll
  for (int kk = 0; kk < DHEAD; kk += 32) {
    v16h a = frag_a(Aq + kk, DHEAD, lane);
#pragma unroll
    for (int j = 0; j < 4; ++j) {
      v16h bf = frag_b(Bk + (size_t)(n0 + 16 * j) * DHEAD + kk, DHEAD, lane);
      acc[j] = wmma_f16(a, bf, acc[j]);
    }
  }
  int mrow = (lane >> 4) << 3;
#pragma unroll
  for (int j = 0; j < 4; ++j)
#pragma unroll
    for (int rr = 0; rr < 8; ++rr)
      Sl[mrow + rr][n0 + 16 * j + (lane & 15)] = acc[j][rr];
  __syncthreads();

  int row = threadIdx.x >> 4;           // 16 threads per row, 32 cols each
  int sub = threadIdx.x & 15;
  int c0  = sub * 32;
  float lm = 0.0f;                      // includes the t=1.0 zero-score point
  for (int c = 0; c < 32; ++c) lm = fmaxf(lm, Sl[row][c0 + c]);
  red[row][sub] = lm;
  __syncthreads();
  if (sub == 0) {
    float m = red[row][0];
    for (int s2 = 1; s2 < 16; ++s2) m = fmaxf(m, red[row][s2]);
    rowM[row] = m;
  }
  __syncthreads();
  float m  = rowM[row];
  float zs = 0.0f;
  for (int c = 0; c < 32; ++c) {
    int n   = c0 + c;
    int cnt = (999 * (n + 1) + 511) / 512 - (999 * n + 511) / 512;
    float wn = DTQ * (float)cnt - (n == 0 ? 0.5f * DTQ : 0.0f);
    float ev = __expf(Sl[row][n] - m) * wn;
    Sl[row][n] = ev;
    zs += ev;
  }
  red[row][sub] = zs;
  __syncthreads();
  if (sub == 0) {
    float z = 0.5f * DTQ * __expf(-m);  // trapezoid endpoint at t=1 (psi==0)
    for (int s2 = 0; s2 < 16; ++s2) z += red[row][s2];
    rowZ[row] = 1.0f / z;
  }
  __syncthreads();
  float iz = rowZ[row];
  _Float16* dst = rmat + ((size_t)bh * TQ + t0 + row) * NBAS;
  for (int c = 0; c < 32; ++c) dst[c0 + c] = (_Float16)(Sl[row][c0 + c] * iz);
}

// ---- GEMM 3: ctx = r @ values ; out[b][t][h*64+d]  (f32) -------------------

__global__ void attn_ctx(const _Float16* __restrict__ rmat,
                         const _Float16* __restrict__ valsT,
                         float* __restrict__ out) {
  int lane = threadIdx.x & 31;
  int wid  = threadIdx.x >> 5;
  int bh = blockIdx.x;                  // 128
  int b = bh >> 3, h = bh & 7;
  int m0 = wid << 5;                    // 32 query rows per wave, full N=64
  const _Float16* A  = rmat  + (size_t)bh * TQ * NBAS;
  const _Float16* Bv = valsT + (size_t)bh * DHEAD * NBAS;
  v8f acc[2][4] = {};
  for (int kk = 0; kk < NBAS; kk += 32) {
    v16h a0 = frag_a(A + (size_t)m0 * NBAS + kk, NBAS, lane);
    v16h a1 = frag_a(A + (size_t)(m0 + 16) * NBAS + kk, NBAS, lane);
#pragma unroll
    for (int j = 0; j < 4; ++j) {
      v16h bf = frag_b(Bv + (size_t)(16 * j) * NBAS + kk, NBAS, lane);
      acc[0][j] = wmma_f16(a0, bf, acc[0][j]);
      acc[1][j] = wmma_f16(a1, bf, acc[1][j]);
    }
  }
#pragma unroll
  for (int i = 0; i < 2; ++i)
#pragma unroll
    for (int j = 0; j < 4; ++j) {
      int d  = 16 * j + (lane & 15);
      int tb = m0 + 16 * i + ((lane >> 4) << 3);
      float* dst = out + ((size_t)b * TQ + tb) * EDIM + h * DHEAD + d;
#pragma unroll
      for (int rr = 0; rr < 8; ++rr)
        dst[(size_t)rr * EDIM] = acc[i][j][rr];
    }
}

// ---------------------------------------------------------------------------

extern "C" void kernel_launch(void* const* d_in, const int* in_sizes, int n_in,
                              void* d_out, int out_size, void* d_ws, size_t ws_size,
                              hipStream_t stream) {
  const float* k  = (const float*)d_in[0];   // [16, 512, 2048]
  const float* q  = (const float*)d_in[1];   // [16, 256, 512]
  const float* Wk = (const float*)d_in[2];   // [512, 512]
  const float* Wv = (const float*)d_in[3];   // [512, 512]
  float* out = (float*)d_out;                // [16, 256, 512]

  char* ws = (char*)d_ws;                    // needs ~64 MB
  _Float16* bc    = (_Float16*)ws;                                   // 8 MB
  _Float16* wkvt  = bc    + (size_t)NBATCH * NBAS * EDIM;            // 1 MB
  _Float16* qh    = wkvt  + (size_t)1024 * EDIM;                     // 4 MB
  _Float16* keys  = qh    + (size_t)NBATCH * NHEAD * TQ * DHEAD;     // 8 MB
  _Float16* valsT = keys  + (size_t)NBATCH * NHEAD * NBAS * DHEAD;   // 8 MB
  _Float16* rmat  = valsT + (size_t)NBATCH * NHEAD * DHEAD * NBAS;   // 32 MB

  prep_bc   <<<2048, 256, 0, stream>>>(k, bc);
  prep_qh   <<<8192, 256, 0, stream>>>(q, qh);
  prep_w    <<<2048, 256, 0, stream>>>(Wk, Wv, wkvt);
  gemm_kv   <<< 512, 256, 0, stream>>>(bc, wkvt, keys, valsT);
  attn_scores<<<2048, 256, 0, stream>>>(qh, keys, rmat);
  attn_ctx  <<< 128, 256, 0, stream>>>(rmat, valsT, out);
}
